// Bengio03biLm_43731357007884
// MI455X (gfx1250) — matrix operand order repaired and checked by reading, not compiled, not run on hardware
//
#include <hip/hip_runtime.h>
#include <stdint.h>

// ---------- types (avoid HIP vector classes inside unions) ----------
typedef __bf16        v16bf  __attribute__((ext_vector_type(16)));
typedef float         v8f    __attribute__((ext_vector_type(8)));
typedef unsigned int  u32x4  __attribute__((ext_vector_type(4)));
typedef unsigned int  u32x2  __attribute__((ext_vector_type(2)));
typedef float         f32x4  __attribute__((ext_vector_type(4)));
typedef int           i32x4  __attribute__((ext_vector_type(4)));

// address-space-qualified pointer element types for the async builtin
typedef __attribute__((address_space(1))) i32x4 gas_i32x4;   // global
typedef __attribute__((address_space(3))) i32x4 las_i32x4;   // LDS

union Frag { u32x4 u[2]; v16bf v; };

// ---------- problem constants ----------
#define B_DIM 16
#define S_DIM 2048
#define H_DIM 512
#define W_DIM 4
#define L_DIM 2
#define ROWS_PAD (S_DIM + 2 * W_DIM)   // 2056
#define C2 (2 * H_DIM)                 // 1024

// ---------- GEMM tiling ----------
#define BM 128
#define BN 64
#define KB 64
#define LDSS 72   // LDS row stride (bf16 elems): 144B = 9*16B keeps 16B alignment

// ---------- CDNA5 async global->LDS ----------
#if __has_builtin(__builtin_amdgcn_global_load_async_to_lds_b128)
#define HAS_ASYNC 1
#else
#define HAS_ASYNC 0
#endif

__device__ __forceinline__ void async_cp16(const unsigned short* g, unsigned short* l) {
#if HAS_ASYNC
  __builtin_amdgcn_global_load_async_to_lds_b128(
      (gas_i32x4*)g, (las_i32x4*)l, 0, 0);
#else
  *(u32x4*)l = *(const u32x4*)g;   // sync fallback
#endif
}

__device__ __forceinline__ void async_wait0() {
#if HAS_ASYNC
#if __has_builtin(__builtin_amdgcn_s_wait_asynccnt)
  __builtin_amdgcn_s_wait_asynccnt(0);
#else
  asm volatile("s_wait_asynccnt 0" ::: "memory");
#endif
#endif
}

__device__ __forceinline__ unsigned short f2bf(float f) {
  unsigned u = __float_as_uint(f);
  u += 0x7FFFu + ((u >> 16) & 1u);
  return (unsigned short)(u >> 16);
}

// ============================================================================
// Generic bf16 WMMA GEMM: C = act(A[MxK] * W[KxN] + bias) (+ residual)
//   A : bf16, row stride lda, per-z stride strideAz
//   Bt: bf16 weights PRE-TRANSPOSED to [N][K] row-major (ld = K)
//   outputs: optional f32 (ldc, per-z strideCz) and/or bf16 (ldcb)
// Grid: (N/BN, M/BM, Z); Block: 256 (8 waves; each wave = 32x32 = 2x2 WMMA)
// Double-buffered LDS, fed by GLOBAL_LOAD_ASYNC_TO_LDS_B128 (ASYNCcnt).
// ============================================================================
__global__ __launch_bounds__(256)
void gemm_bf16_wmma(const unsigned short* __restrict__ A, int lda, long long strideAz,
                    const unsigned short* __restrict__ Bt,
                    const float* __restrict__ bias,
                    const float* __restrict__ Res, int ldr, long long strideRz,
                    float* __restrict__ Cf, int ldc, long long strideCz,
                    unsigned short* __restrict__ Cb, int ldcb,
                    int K, int relu)
{
  __shared__ unsigned short As[2][BM * LDSS];   // [row m][col k]
  __shared__ unsigned short Bs[2][BN * LDSS];   // [row n][col k] (weights already N-major)

  const int tid  = threadIdx.x;
  const int lane = tid & 31;
  const int wave = tid >> 5;
  const int wm   = (wave >> 1) * 32;            // 0,32,64,96
  const int wn   = (wave & 1) * 32;             // 0,32

  const int z  = blockIdx.z;
  const int m0 = blockIdx.y * BM;
  const int n0 = blockIdx.x * BN;

  const unsigned short* Ab = A + (long long)z * strideAz + (long long)m0 * lda;

  v8f acc[2][2];
  #pragma unroll
  for (int i = 0; i < 2; ++i)
    #pragma unroll
    for (int j = 0; j < 2; ++j)
      #pragma unroll
      for (int r = 0; r < 8; ++r) acc[i][j][r] = 0.0f;

  // per-thread copy slots (A: 4 x B128, B: 2 x B128 per K-chunk)
  const int aR[4]  = { (tid + 0*256) >> 3, (tid + 1*256) >> 3, (tid + 2*256) >> 3, (tid + 3*256) >> 3 };
  const int aCG    = (tid & 7) * 8;
  const int bR[2]  = { (tid + 0*256) >> 3, (tid + 1*256) >> 3 };

  auto issue = [&](int k0, int p) {
    #pragma unroll
    for (int i = 0; i < 4; ++i)
      async_cp16(Ab + (long long)aR[i] * lda + k0 + aCG, &As[p][aR[i] * LDSS + aCG]);
    #pragma unroll
    for (int i = 0; i < 2; ++i)
      async_cp16(Bt + (long long)(n0 + bR[i]) * K + k0 + aCG, &Bs[p][bR[i] * LDSS + aCG]);
  };

  issue(0, 0);
  const int nk = K / KB;
  for (int kt = 0; kt < nk; ++kt) {
    const int p = kt & 1;
    async_wait0();        // chunk kt finished landing in LDS
    __syncthreads();
    if (kt + 1 < nk) issue((kt + 1) * KB, p ^ 1);

    // two 16x16x32 K-steps per staged chunk
    #pragma unroll
    for (int ks = 0; ks < 2; ++ks) {
      const int kb = ks * 32;
      const int hA = (lane >> 4) * 8;    // A: lanes 16-31 hold K+8..15 / K+24..31
      const int hB = (lane >> 4) * 8;    // B: lanes 16-31 hold K+8..15 / K+24..31
      Frag fa[2], fb[2];
      #pragma unroll
      for (int i = 0; i < 2; ++i) {
        const unsigned short* pa = &As[p][(wm + i * 16 + (lane & 15)) * LDSS + kb + hA];
        fa[i].u[0] = *(const u32x4*)(pa);
        fa[i].u[1] = *(const u32x4*)(pa + 16);
      }
      #pragma unroll
      for (int j = 0; j < 2; ++j) {
        const unsigned short* pb = &Bs[p][(wn + j * 16 + (lane & 15)) * LDSS + kb + hB];
        fb[j].u[0] = *(const u32x4*)(pb);
        fb[j].u[1] = *(const u32x4*)(pb + 16);
      }
      #pragma unroll
      for (int i = 0; i < 2; ++i)
        #pragma unroll
        for (int j = 0; j < 2; ++j)
          acc[i][j] = __builtin_amdgcn_wmma_f32_16x16x32_bf16(
              false, fa[i].v, false, fb[j].v, (short)0, acc[i][j], false, false);
    }
  }

  // epilogue: C VGPR r holds (M = base + r + 8*(lane>=16), N = base + lane&15)
  const int colL = lane & 15;
  const int rh   = (lane >> 4) * 8;
  #pragma unroll
  for (int i = 0; i < 2; ++i) {
    #pragma unroll
    for (int j = 0; j < 2; ++j) {
      const int gn = n0 + wn + j * 16 + colL;
      const float bv = bias ? bias[gn] : 0.0f;
      #pragma unroll
      for (int r = 0; r < 8; ++r) {
        const long long gm = (long long)m0 + wm + i * 16 + rh + r;
        float v = acc[i][j][r] + bv;
        if (relu) v = v > 0.0f ? v : 0.0f;
        if (Res) v += Res[(long long)z * strideRz + gm * ldr + gn];
        if (Cf) Cf[(long long)z * strideCz + gm * ldc + gn] = v;
        if (Cb) Cb[gm * ldcb + gn] = f2bf(v);
      }
    }
  }
}

// ============================================================================
// Wave-per-row LayerNorm over H=512 f32 (row stride ldx) -> bf16 [row*512]
// ============================================================================
__global__ __launch_bounds__(256)
void ln_bf16(const float* __restrict__ X, int ldx,
             const float* __restrict__ g, const float* __restrict__ be,
             unsigned short* __restrict__ out)
{
  const int wave = threadIdx.x >> 5, lane = threadIdx.x & 31;
  const long long row = (long long)blockIdx.x * 8 + wave;
  const float* xr = X + row * ldx + lane * 16;

  float x[16];
  #pragma unroll
  for (int q = 0; q < 4; ++q) {
    f32x4 v = *(const f32x4*)(xr + q * 4);
    x[q * 4 + 0] = v.x; x[q * 4 + 1] = v.y; x[q * 4 + 2] = v.z; x[q * 4 + 3] = v.w;
  }
  float s = 0.0f, qq = 0.0f;
  #pragma unroll
  for (int e = 0; e < 16; ++e) { s += x[e]; qq += x[e] * x[e]; }
  #pragma unroll
  for (int m = 16; m > 0; m >>= 1) { s += __shfl_xor(s, m); qq += __shfl_xor(qq, m); }

  const float mean = s * (1.0f / 512.0f);
  const float var  = qq * (1.0f / 512.0f) - mean * mean;
  const float inv  = rsqrtf(var + 1e-5f);

  alignas(16) unsigned short o[16];
  const int base = lane * 16;
  #pragma unroll
  for (int e = 0; e < 16; ++e)
    o[e] = f2bf(g[base + e] * (x[e] - mean) * inv + be[base + e]);
  unsigned short* dst = out + row * H_DIM + base;
  *(u32x4*)(dst)     = ((const u32x4*)o)[0];
  *(u32x4*)(dst + 8) = ((const u32x4*)o)[1];
}

// ============================================================================
// Weight transpose + convert: in f32 [L][K][N] -> out bf16 [L][N][K]
// ============================================================================
__global__ __launch_bounds__(256)
void transpose_w_bf16(const float* __restrict__ in, unsigned short* __restrict__ out,
                      int K, int N, long long total)
{
  long long idx = (long long)blockIdx.x * blockDim.x + threadIdx.x;
  if (idx >= total) return;
  const long long mn  = (long long)K * N;
  const long long mat = idx / mn;
  const int r = (int)(idx % mn);
  const int k = r / N, n = r % N;
  out[mat * mn + (long long)n * K + k] = f2bf(in[idx]);
}

// ============================================================================
// Build bf16 padded buffer [B][S+2W][H] = [left pad | inputs | right pad]
// ============================================================================
__global__ __launch_bounds__(256)
void build_pad(const float* __restrict__ inp, const float* __restrict__ lp,
               const float* __restrict__ rp, unsigned short* __restrict__ pad)
{
  const long long total = (long long)B_DIM * ROWS_PAD * (H_DIM / 4);
  long long idx = (long long)blockIdx.x * blockDim.x + threadIdx.x;
  if (idx >= total) return;
  const int c  = (int)(idx % (H_DIM / 4)) * 4;
  const long long rg = idx / (H_DIM / 4);
  const int r = (int)(rg % ROWS_PAD);
  const int b = (int)(rg / ROWS_PAD);
  const float* src;
  if (r < W_DIM)              src = lp + (long long)r * H_DIM + c;
  else if (r < W_DIM + S_DIM) src = inp + ((long long)b * S_DIM + (r - W_DIM)) * H_DIM + c;
  else                        src = rp + (long long)(r - W_DIM - S_DIM) * H_DIM + c;
  f32x4 v = *(const f32x4*)src;
  alignas(8) unsigned short o[4] = { f2bf(v.x), f2bf(v.y), f2bf(v.z), f2bf(v.w) };
  *(u32x2*)(pad + rg * H_DIM + c) = *(const u32x2*)o;
}

// ============================================================================
// Snapshot: X [B][S][2H] -> all_layers[l] layout [S][B][2H]
// ============================================================================
__global__ __launch_bounds__(256)
void snapshot(const float* __restrict__ X, float* __restrict__ outL)
{
  long long idx = ((long long)blockIdx.x * blockDim.x + threadIdx.x) * 4;
  const long long total = (long long)S_DIM * B_DIM * C2;
  if (idx >= total) return;
  const long long c = idx % C2;
  const long long t = idx / C2;
  const long long b = t % B_DIM;
  const long long s = t / B_DIM;
  f32x4 v = *(const f32x4*)(X + (b * S_DIM + s) * C2 + c);
  *(f32x4*)(outL + (s * B_DIM + b) * C2 + c) = v;
}

// ============================================================================
extern "C" void kernel_launch(void* const* d_in, const int* in_sizes, int n_in,
                              void* d_out, int out_size, void* d_ws, size_t ws_size,
                              hipStream_t stream)
{
  const float* inputs = (const float*)d_in[0];
  const float* lpad   = (const float*)d_in[1];
  const float* rpad   = (const float*)d_in[2];
  const float* Wl     = (const float*)d_in[3];
  const float* bl     = (const float*)d_in[4];
  const float* Wr     = (const float*)d_in[5];
  const float* br     = (const float*)d_in[6];
  const float* lw1    = (const float*)d_in[7];
  const float* lb1    = (const float*)d_in[8];
  const float* lw2    = (const float*)d_in[9];
  const float* lb2    = (const float*)d_in[10];
  const float* lg     = (const float*)d_in[11];
  const float* lbe    = (const float*)d_in[12];
  const float* rw1    = (const float*)d_in[13];
  const float* rb1    = (const float*)d_in[14];
  const float* rw2    = (const float*)d_in[15];
  const float* rb2    = (const float*)d_in[16];
  const float* rg     = (const float*)d_in[17];
  const float* rbe    = (const float*)d_in[18];

  float* out = (float*)d_out;
  float* X   = out + (long long)L_DIM * S_DIM * B_DIM * C2;   // last_layers region [B][S][2H]

  // ---- workspace carve-up ----
  char* ws = (char*)d_ws;
  size_t off = 0;
  auto carve = [&](size_t bytes) -> void* {
    void* p = ws + off;
    off += (bytes + 255) & ~(size_t)255;
    return p;
  };
  const long long nWproj = (long long)W_DIM * H_DIM * H_DIM;   // 2048*512
  const long long nWff   = (long long)L_DIM * H_DIM * H_DIM;   // 2*512*512
  const long long nTok   = (long long)B_DIM * S_DIM;           // 32768

  unsigned short* pad_bf = (unsigned short*)carve((size_t)B_DIM * ROWS_PAD * H_DIM * 2);
  unsigned short* wlT    = (unsigned short*)carve((size_t)nWproj * 2);  // [N=512][K=2048]
  unsigned short* wrT    = (unsigned short*)carve((size_t)nWproj * 2);
  unsigned short* lw1T   = (unsigned short*)carve((size_t)nWff * 2);   // [L][N=512][K=512]
  unsigned short* lw2T   = (unsigned short*)carve((size_t)nWff * 2);
  unsigned short* rw1T   = (unsigned short*)carve((size_t)nWff * 2);
  unsigned short* rw2T   = (unsigned short*)carve((size_t)nWff * 2);
  unsigned short* xln_bf = (unsigned short*)carve((size_t)nTok * H_DIM * 2);
  unsigned short* h_bf   = (unsigned short*)carve((size_t)nTok * H_DIM * 2);

  // ---- 1. weight transpose+convert f32 [K][N] -> bf16 [N][K] ----
  auto tw = [&](const float* src, unsigned short* dst, int K, int N, long long L) {
    long long total = L * (long long)K * N;
    transpose_w_bf16<<<(int)((total + 255) / 256), 256, 0, stream>>>(src, dst, K, N, total);
  };
  tw(Wl,  wlT,  W_DIM * H_DIM, H_DIM, 1);
  tw(Wr,  wrT,  W_DIM * H_DIM, H_DIM, 1);
  tw(lw1, lw1T, H_DIM, H_DIM, L_DIM);
  tw(lw2, lw2T, H_DIM, H_DIM, L_DIM);
  tw(rw1, rw1T, H_DIM, H_DIM, L_DIM);
  tw(rw2, rw2T, H_DIM, H_DIM, L_DIM);

  // ---- 2. padded bf16 activation buffer ----
  {
    long long total = (long long)B_DIM * ROWS_PAD * (H_DIM / 4);
    build_pad<<<(int)((total + 255) / 256), 256, 0, stream>>>(inputs, lpad, rpad, pad_bf);
  }

  // ---- 3. context projections (overlapping-row GEMM, lda = H) ----
  // left:  row t of A = padded rows [t, t+W)        -> base offset 0
  // right: row t of A = padded rows [t+W+1, t+2W+1) -> base offset (W+1)*H
  {
    dim3 grid(H_DIM / BN, S_DIM / BM, B_DIM);
    gemm_bf16_wmma<<<grid, 256, 0, stream>>>(
        pad_bf, H_DIM, (long long)ROWS_PAD * H_DIM,
        wlT, bl,
        nullptr, 0, 0,
        X, C2, (long long)S_DIM * C2,
        nullptr, 0,
        W_DIM * H_DIM, /*relu=*/1);
    gemm_bf16_wmma<<<grid, 256, 0, stream>>>(
        pad_bf + (W_DIM + 1) * H_DIM, H_DIM, (long long)ROWS_PAD * H_DIM,
        wrT, br,
        nullptr, 0, 0,
        X + H_DIM, C2, (long long)S_DIM * C2,
        nullptr, 0,
        W_DIM * H_DIM, /*relu=*/1);
  }

  // ---- 4. residual FF layers, both sides, in place in X ----
  for (int l = 0; l < L_DIM; ++l) {
    for (int side = 0; side < 2; ++side) {
      const int colOff = side * H_DIM;
      const unsigned short* w1b = (side ? rw1T : lw1T) + (long long)l * H_DIM * H_DIM;
      const unsigned short* w2b = (side ? rw2T : lw2T) + (long long)l * H_DIM * H_DIM;
      const float* b1 = (side ? rb1 : lb1) + (long long)l * H_DIM;
      const float* b2 = (side ? rb2 : lb2) + (long long)l * H_DIM;
      const float* gg = (side ? rg  : lg)  + (long long)l * H_DIM;
      const float* bb = (side ? rbe : lbe) + (long long)l * H_DIM;

      ln_bf16<<<(int)(nTok / 8), 256, 0, stream>>>(X + colOff, C2, gg, bb, xln_bf);

      dim3 grid(H_DIM / BN, (int)(nTok / BM), 1);
      // h = relu(ln(x) @ W1 + b1)   (bf16 out only)
      gemm_bf16_wmma<<<grid, 256, 0, stream>>>(
          xln_bf, H_DIM, 0,
          w1b, b1,
          nullptr, 0, 0,
          nullptr, 0, 0,
          h_bf, H_DIM,
          H_DIM, /*relu=*/1);
      // x = x + h @ W2 + b2         (f32, in place)
      gemm_bf16_wmma<<<grid, 256, 0, stream>>>(
          h_bf, H_DIM, 0,
          w2b, b2,
          X + colOff, C2, 0,
          X + colOff, C2, 0,
          nullptr, 0,
          H_DIM, /*relu=*/0);
    }
    // all_layers[l] : [S][B][2H] <- X [B][S][2H]
    long long total = (long long)S_DIM * B_DIM * C2 / 4;
    snapshot<<<(int)((total + 255) / 256), 256, 0, stream>>>(
        X, out + (long long)l * S_DIM * B_DIM * C2);
  }
  // X already equals last_layers (it lives in d_out) -> done.
  (void)in_sizes; (void)n_in; (void)out_size; (void)ws_size;
}
